// Attention_6287832122019
// MI455X (gfx1250) — matrix operand order repaired and checked by reading, not compiled, not run on hardware
//
#include <hip/hip_runtime.h>
#include <math.h>

// ---------------------------------------------------------------------------
// Problem constants (B=2, S=2048, D=2048, H=16, DH=128)
// ---------------------------------------------------------------------------
#define Bb   2
#define Ss   2048
#define Dd   2048
#define Hh   16
#define DHd  128
#define Mdim (Bb * Ss)      // 4096
#define Ndim (Hh * DHd)     // 2048
#define Kdim Dd             // 2048
#define HEAD_ELEMS ((size_t)Bb * Hh * Ss * DHd)  // 8,388,608 elements per buffer

// ---------------------------------------------------------------------------
// bf16 helpers (round-to-nearest-even) + WMMA fragment types
// ---------------------------------------------------------------------------
typedef __bf16 bf16x16 __attribute__((ext_vector_type(16)));
typedef float  f32x8   __attribute__((ext_vector_type(8)));

union Frag { bf16x16 bf; unsigned int u[8]; };

__device__ inline unsigned short f2bf(float f) {
    unsigned int u = __float_as_uint(f);
    unsigned int r = u + 0x7FFFu + ((u >> 16) & 1u);
    return (unsigned short)(r >> 16);
}
__device__ inline unsigned int pack_bf2(float lo, float hi) {
    return (unsigned int)f2bf(lo) | ((unsigned int)f2bf(hi) << 16);
}
__device__ inline float bf2f(unsigned short h) {
    return __uint_as_float(((unsigned int)h) << 16);
}

// ---------------------------------------------------------------------------
// Tensor Data Mover: 2D tile load (rows x 128 bf16 elems, contiguous rows)
// D# per CDNA5 ISA ch.8: group0 {count, lds_addr, global_addr, type=2},
// group1 {data_size=2B, tensor_dim0=128, tensor_dim1=S, tile 128 x rows,
// tensor_dim0_stride=128}. Issued once per wave (TDM ignores EXEC); tracked
// with TENSORcnt.
// ---------------------------------------------------------------------------
#define HAS_TDM __has_builtin(__builtin_amdgcn_tensor_load_to_lds)

#if HAS_TDM
typedef unsigned int u32x4 __attribute__((ext_vector_type(4)));
typedef int          i32x4 __attribute__((ext_vector_type(4)));
typedef int          i32x8 __attribute__((ext_vector_type(8)));

__device__ inline void tdm_load_tile_2d(const unsigned short* gsrc,
                                        unsigned short* lds_dst, int rows)
{
    const unsigned long long ga = (unsigned long long)(uintptr_t)gsrc;
    const unsigned int lds_off  = (unsigned int)(uintptr_t)lds_dst; // low 32b = LDS offset
    u32x4 g0;
    g0.x = 1u;                                              // count=1, user descriptor
    g0.y = lds_off;                                         // lds_addr
    g0.z = (unsigned int)(ga & 0xFFFFFFFFu);                // global_addr[31:0]
    g0.w = (unsigned int)((ga >> 32) & 0x01FFFFFFu)         // global_addr[56:32]
         | (2u << 30);                                      // type = 2 ("image")
    i32x8 g1;
    g1[0] = (int)(1u << 16);                                // data_size=1 -> 2 bytes
    g1[1] = (int)(128u << 16);                              // tensor_dim0[15:0] = 128
    g1[2] = (int)(((unsigned)Ss & 0xFFFFu) << 16);          // tensor_dim1[15:0] = S
    g1[3] = (int)(128u << 16);                              // tile_dim0 = 128
    g1[4] = rows;                                           // tile_dim1 = rows
    g1[5] = 128;                                            // tensor_dim0_stride = 128
    g1[6] = 0;
    g1[7] = 0;
    const i32x4 z4 = {0, 0, 0, 0};
#if __clang_major__ >= 23
    const i32x8 z8 = {0, 0, 0, 0, 0, 0, 0, 0};
    __builtin_amdgcn_tensor_load_to_lds(g0, g1, z4, z4, z8, 0);
#else
    __builtin_amdgcn_tensor_load_to_lds(g0, g1, z4, z4, 0);
#endif
}
#endif // HAS_TDM

// ---------------------------------------------------------------------------
// GEMM kernel: C[M,N] = A[M,K] * W[K,N] + bias[N]
//   MODE 0 (QKV proj): A = f32 activations [4096,2048] row-major,
//                      W = f32 head-blocked (H,D,DH), out = bf16 ws [B,H,S,DH]
//   MODE 1 (O proj):   A = bf16 ws z [B,H,S,DH] viewed as [4096,2048],
//                      W = f32 (H,DH,D) == row-major [2048,2048], out = f32 d_out
// Block: 256 threads (8 waves). Tile: BM=128, BN=128, BK=32.
// Wave grid 2(M)x4(N); each wave computes 64x32 via 4x2 WMMA accumulators.
// ---------------------------------------------------------------------------
template <int MODE>
__global__ __launch_bounds__(256)
void gemm_wmma(const void* __restrict__ Aptr, const float* __restrict__ W,
               const float* __restrict__ bias, void* __restrict__ Out)
{
    __shared__ unsigned short As[128 * 32];   // A tile, row-major [m][k], bf16
    __shared__ unsigned short Bs[128 * 32];   // W tile transposed: Bt[n][k], bf16

    const int tid  = threadIdx.x;
    const int lane = tid & 31;
    const int w    = tid >> 5;
    const int wm   = w & 1;     // 0..1  (M direction)
    const int wn   = w >> 1;    // 0..3  (N direction)
    const int m0   = blockIdx.x * 128;
    const int n0   = blockIdx.y * 128;
    const int half = lane >> 4;
    const int mn   = lane & 15;

    f32x8 acc[4][2] = {};

    for (int kt = 0; kt < Kdim / 32; ++kt) {
        const int k0 = kt * 32;
        __syncthreads();
        // ---- stage A tile: thread -> row = tid>>1, 16 consecutive k ----
        {
            const int row = tid >> 1;
            const int cb  = (tid & 1) * 16;
            const int gm  = m0 + row;
            if (MODE == 0) {
                const float* Af = (const float*)Aptr;
                const float4* src = (const float4*)(Af + (size_t)gm * Kdim + k0 + cb);
                if (kt + 1 < Kdim / 32)                 // global_prefetch_b8
                    __builtin_prefetch(src + 8, 0, 1);  // next k-tile, same row
                unsigned int* dst = (unsigned int*)&As[row * 32 + cb];
                #pragma unroll
                for (int i = 0; i < 4; ++i) {
                    float4 f = src[i];
                    dst[i * 2 + 0] = pack_bf2(f.x, f.y);
                    dst[i * 2 + 1] = pack_bf2(f.z, f.w);
                }
            } else {
                const unsigned short* Ab = (const unsigned short*)Aptr;
                const int bb = gm >> 11;        // batch
                const int s  = gm & (Ss - 1);   // seq pos
                #pragma unroll
                for (int i = 0; i < 16; i += 2) {
                    const int k = k0 + cb + i;
                    const int h = k >> 7, e = k & 127;  // pair stays within head
                    const unsigned int p = *(const unsigned int*)
                        &Ab[(((size_t)bb * Hh + h) * Ss + s) * DHd + e];
                    *(unsigned int*)&As[row * 32 + cb + i] = p;
                }
            }
        }
        // ---- stage W tile transposed: thread -> nl = tid>>1, 16 k values ----
        {
            const int nl = tid >> 1;
            const int kb = (tid & 1) * 16;
            #pragma unroll
            for (int i = 0; i < 16; i += 2) {
                const int k = k0 + kb + i;
                float x, y;
                if (MODE == 0) {
                    // W (H,D,DH): off = head*D*DH + k*DH + e ; n0 multiple of 128
                    const size_t base = (size_t)(n0 >> 7) * ((size_t)Kdim * DHd) + (size_t)nl;
                    x = W[base + (size_t)k * DHd];
                    y = W[base + (size_t)(k + 1) * DHd];
                    if (i == 0 && kt + 1 < Kdim / 32)
                        __builtin_prefetch(&W[base + (size_t)(k + 32) * DHd], 0, 1);
                } else {
                    // W_O (H,DH,D) flattens to row-major [K=2048][N=2048]
                    x = W[(size_t)k * Ndim + n0 + nl];
                    y = W[(size_t)(k + 1) * Ndim + n0 + nl];
                    if (i == 0 && kt + 1 < Kdim / 32)
                        __builtin_prefetch(&W[(size_t)(k + 32) * Ndim + n0 + nl], 0, 1);
                }
                *(unsigned int*)&Bs[nl * 32 + kb + i] = pack_bf2(x, y);
            }
        }
        __syncthreads();
        // ---- load fragments per documented 16-bit A/B layouts ----
        Frag a[4], bfr[2];
        #pragma unroll
        for (int mf = 0; mf < 4; ++mf) {
            const int row = wm * 64 + mf * 16 + mn;
            #pragma unroll
            for (int v = 0; v < 8; ++v) {
                const int grp = v >> 2, vv = v & 3;
                const int k = grp * 16 + half * 8 + 2 * vv;
                a[mf].u[v] = *(const unsigned int*)&As[row * 32 + k];
            }
        }
        #pragma unroll
        for (int nf = 0; nf < 2; ++nf) {
            const int row = wn * 32 + nf * 16 + mn;
            #pragma unroll
            for (int v = 0; v < 8; ++v) {
                const int k = (v >> 2) * 16 + half * 8 + 2 * (v & 3);
                bfr[nf].u[v] = *(const unsigned int*)&Bs[row * 32 + k];
            }
        }
        #pragma unroll
        for (int mf = 0; mf < 4; ++mf)
            #pragma unroll
            for (int nf = 0; nf < 2; ++nf)
                acc[mf][nf] = __builtin_amdgcn_wmma_f32_16x16x32_bf16(
                    false, a[mf].bf, false, bfr[nf].bf,
                    (short)0, acc[mf][nf], false, false);
    }

    // ---- epilogue: +bias, scatter per C-matrix layout (m = r+8*half, n = lane&15)
    #pragma unroll
    for (int mf = 0; mf < 4; ++mf) {
        #pragma unroll
        for (int nf = 0; nf < 2; ++nf) {
            #pragma unroll
            for (int r = 0; r < 8; ++r) {
                const int gm = m0 + wm * 64 + mf * 16 + r + 8 * half;
                const int gn = n0 + wn * 32 + nf * 16 + mn;
                const float vout = acc[mf][nf][r] + bias[gn];
                if (MODE == 0) {
                    const int bb = gm >> 11, s = gm & (Ss - 1);
                    const int h = gn >> 7, e = gn & 127;
                    ((unsigned short*)Out)[(((size_t)bb * Hh + h) * Ss + s) * DHd + e] =
                        f2bf(vout);
                } else {
                    ((float*)Out)[(size_t)gm * Ndim + gn] = vout;
                }
            }
        }
    }
}

// ---------------------------------------------------------------------------
// Rotary embedding, in place on bf16 Q and K [B,H,S,DH].
// Pair (e, e+64) shares angle = pos * base^(-e/64); one thread per pair.
// ---------------------------------------------------------------------------
__global__ __launch_bounds__(256)
void rotary_kernel(unsigned short* __restrict__ Q, unsigned short* __restrict__ K)
{
    const size_t idx = (size_t)blockIdx.x * blockDim.x + threadIdx.x;
    const int e = (int)(idx & 63);
    const size_t row = idx >> 6;               // b*H*S + h*S + s
    const int s = (int)(row & (Ss - 1));
    const float inv_freq = __powf(10000.0f, -((float)e) * (1.0f / 64.0f));
    const float ang = (float)s * inv_freq;
    float sn, c;
    __sincosf(ang, &sn, &c);
    const size_t base = row * DHd;

    const float x0 = bf2f(Q[base + e]), x1 = bf2f(Q[base + e + 64]);
    Q[base + e]      = f2bf(x0 * c - x1 * sn);
    Q[base + e + 64] = f2bf(x1 * c + x0 * sn);

    const float y0 = bf2f(K[base + e]), y1 = bf2f(K[base + e + 64]);
    K[base + e]      = f2bf(y0 * c - y1 * sn);
    K[base + e + 64] = f2bf(y1 * c + y0 * sn);
}

// ---------------------------------------------------------------------------
// Flash attention (causal, online softmax).
// Grid: (S/64, H, B). Block: 128 threads = 4 waves; wave w owns q rows
// [blk*64 + w*16, +16). 32-key tiles staged via TDM (tensor_load_to_lds).
// ---------------------------------------------------------------------------
__global__ __launch_bounds__(128)
void flash_attn(const unsigned short* __restrict__ Q,
                const unsigned short* __restrict__ K,
                const unsigned short* __restrict__ V,
                unsigned short* __restrict__ Z)
{
    const int qblk = blockIdx.x, h = blockIdx.y, b = blockIdx.z;
    const int tid = threadIdx.x, lane = tid & 31, w = tid >> 5;
    const int half = lane >> 4, mn = lane & 15;
    const size_t headoff = ((size_t)b * Hh + h) * (size_t)Ss * DHd;
    const unsigned short* Qh = Q + headoff;
    const unsigned short* Kh = K + headoff;
    const unsigned short* Vh = V + headoff;
    unsigned short* Zh = Z + headoff;

    __shared__ unsigned short k_s[32 * 128];      // K tile rows (== Bt for Q.K^T)
    __shared__ unsigned short v_s[32 * 128];      // V tile row-major (TDM dest)
    __shared__ unsigned short vt_s[128 * 32];     // V tile transposed: [dh][key]
    __shared__ float          s_f32[4][16 * 32];  // per-wave raw scores
    __shared__ unsigned short p_s[4][16 * 32];    // per-wave probs (bf16, A layout src)
    __shared__ float m_s[4][16], l_s[4][16], a_s[4][16];

    const int qrow0 = qblk * 64 + w * 16;

    // Q A-fragments (16 rows x 128 dh) held in registers for the whole kernel.
    Frag aq[4];
    {
        const unsigned short* qr = Qh + (size_t)(qrow0 + mn) * DHd;
        #pragma unroll
        for (int c = 0; c < 4; ++c)
            #pragma unroll
            for (int v = 0; v < 8; ++v) {
                const int grp = v >> 2, vv = v & 3;
                const int k = c * 32 + grp * 16 + half * 8 + 2 * vv;
                aq[c].u[v] = *(const unsigned int*)&qr[k];
            }
    }

    if (lane < 16) { m_s[w][lane] = -INFINITY; l_s[w][lane] = 0.0f; }
    f32x8 zacc[8] = {};

    const float scale = 0.08838834764831845f;   // 1/sqrt(128)
    const int ntiles = qblk * 2 + 2;            // causal: tiles with key0 <= blk max q

    for (int t = 0; t < ntiles; ++t) {
        const int key0 = t * 32;
        __syncthreads();
        // ---- stage K and V tiles (32x128 bf16 each) ----
#if HAS_TDM
        if (w == 0) {
            tdm_load_tile_2d(Kh + (size_t)key0 * DHd, k_s, 32);
            tdm_load_tile_2d(Vh + (size_t)key0 * DHd, v_s, 32);
            __builtin_amdgcn_s_wait_tensorcnt(0);
        }
#else
        {
            const unsigned int* srcK = (const unsigned int*)(Kh + (size_t)key0 * DHd);
            const unsigned int* srcV = (const unsigned int*)(Vh + (size_t)key0 * DHd);
            unsigned int* dK = (unsigned int*)k_s;
            unsigned int* dV = (unsigned int*)v_s;
            for (int i = tid; i < 2048; i += 128) { dK[i] = srcK[i]; dV[i] = srcV[i]; }
        }
#endif
        if (t + 1 < ntiles) {   // prefetch next K/V tiles: 64B per thread covers 8KB
            __builtin_prefetch(Kh + (size_t)(key0 + 32) * DHd + tid * 32, 0, 1);
            __builtin_prefetch(Vh + (size_t)(key0 + 32) * DHd + tid * 32, 0, 1);
        }
        __syncthreads();
        // ---- transpose V within LDS: vt_s[dh][key] ----
        for (int i = tid; i < 4096; i += 128)
            vt_s[(i & 127) * 32 + (i >> 7)] = v_s[i];
        __syncthreads();
        // ---- scores: S = Q (16x128) . K^T (128x32), two 16-key halves ----
        #pragma unroll
        for (int nsub = 0; nsub < 2; ++nsub) {
            f32x8 sacc = {};
            #pragma unroll
            for (int c = 0; c < 4; ++c) {
                Frag bk;
                #pragma unroll
                for (int v = 0; v < 8; ++v) {
                    const int kk = c * 32 + (v >> 2) * 16 + half * 8 + 2 * (v & 3);
                    bk.u[v] = *(const unsigned int*)&k_s[(nsub * 16 + mn) * 128 + kk];
                }
                sacc = __builtin_amdgcn_wmma_f32_16x16x32_bf16(
                    false, aq[c].bf, false, bk.bf, (short)0, sacc, false, false);
            }
            #pragma unroll
            for (int r = 0; r < 8; ++r)
                s_f32[w][(r + 8 * half) * 32 + nsub * 16 + mn] = sacc[r] * scale;
        }
        __syncthreads();
        // ---- online softmax: lanes 0..15 each own one q row ----
        if (lane < 16) {
            const int m = lane;
            const int qg = qrow0 + m;
            const float mold = m_s[w][m];
            float sv[32];
            float tmax = -INFINITY;
            #pragma unroll
            for (int j = 0; j < 32; ++j) {
                float x = s_f32[w][m * 32 + j];
                if (key0 + j > qg) x = -INFINITY;
                sv[j] = x;
                tmax = fmaxf(tmax, x);
            }
            const float mnew = fmaxf(mold, tmax);
            const float alpha = (mnew == -INFINITY) ? 1.0f : __expf(mold - mnew);
            float lsum = 0.0f;
            #pragma unroll
            for (int j = 0; j < 32; ++j) {
                const float p = (sv[j] == -INFINITY) ? 0.0f : __expf(sv[j] - mnew);
                lsum += p;
                p_s[w][m * 32 + j] = f2bf(p);
            }
            m_s[w][m] = mnew;
            l_s[w][m] = l_s[w][m] * alpha + lsum;
            a_s[w][m] = alpha;
        }
        __syncthreads();
        // ---- rescale running output and accumulate P (16x32) . V (32x128) ----
        Frag ap;
        #pragma unroll
        for (int v = 0; v < 8; ++v) {
            const int grp = v >> 2, vv = v & 3;
            const int k = grp * 16 + half * 8 + 2 * vv;
            ap.u[v] = *(const unsigned int*)&p_s[w][mn * 32 + k];
        }
        float al[8];
        #pragma unroll
        for (int r = 0; r < 8; ++r) al[r] = a_s[w][r + 8 * half];
        #pragma unroll
        for (int d = 0; d < 8; ++d) {
            #pragma unroll
            for (int r = 0; r < 8; ++r) zacc[d][r] *= al[r];
            Frag bv;
            #pragma unroll
            for (int v = 0; v < 8; ++v) {
                const int kk = (v >> 2) * 16 + half * 8 + 2 * (v & 3);
                bv.u[v] = *(const unsigned int*)&vt_s[(d * 16 + mn) * 32 + kk];
            }
            zacc[d] = __builtin_amdgcn_wmma_f32_16x16x32_bf16(
                false, ap.bf, false, bv.bf, (short)0, zacc[d], false, false);
        }
    }
    __syncthreads();
    // ---- finalize: divide by l, write bf16 Z [B,H,S,DH] ----
    float linv[8];
    #pragma unroll
    for (int r = 0; r < 8; ++r) {
        const float l = l_s[w][r + 8 * half];
        linv[r] = (l > 0.0f) ? 1.0f / l : 0.0f;
    }
    #pragma unroll
    for (int d = 0; d < 8; ++d)
        #pragma unroll
        for (int r = 0; r < 8; ++r) {
            const int qg = qrow0 + r + 8 * half;
            const int e = d * 16 + mn;
            Zh[(size_t)qg * DHd + e] = f2bf(zacc[d][r] * linv[r]);
        }
}

// ---------------------------------------------------------------------------
// Host launcher
// ---------------------------------------------------------------------------
extern "C" void kernel_launch(void* const* d_in, const int* in_sizes, int n_in,
                              void* d_out, int out_size, void* d_ws, size_t ws_size,
                              hipStream_t stream)
{
    (void)in_sizes; (void)n_in; (void)out_size; (void)ws_size;
    const float* qin = (const float*)d_in[0];
    const float* kin = (const float*)d_in[1];
    const float* vin = (const float*)d_in[2];
    const float* WQ  = (const float*)d_in[3];
    const float* WK  = (const float*)d_in[4];
    const float* WV  = (const float*)d_in[5];
    const float* WO  = (const float*)d_in[6];
    const float* bQ  = (const float*)d_in[7];
    const float* bK  = (const float*)d_in[8];
    const float* bV  = (const float*)d_in[9];
    const float* bO  = (const float*)d_in[10];

    unsigned short* Qws = (unsigned short*)d_ws;
    unsigned short* Kws = Qws + HEAD_ELEMS;
    unsigned short* Vws = Kws + HEAD_ELEMS;
    unsigned short* Zws = Vws + HEAD_ELEMS;

    const dim3 ggrid(Mdim / 128, Ndim / 128);   // (32, 16)

    // QKV projections (f32 -> bf16 WMMA GEMM, bias fused)
    gemm_wmma<0><<<ggrid, 256, 0, stream>>>(qin, WQ, bQ, Qws);
    gemm_wmma<0><<<ggrid, 256, 0, stream>>>(kin, WK, bK, Kws);
    gemm_wmma<0><<<ggrid, 256, 0, stream>>>(vin, WV, bV, Vws);

    // Rotary on Q and K (one thread per (row, pair))
    const size_t npairs = (size_t)Bb * Hh * Ss * 64;
    rotary_kernel<<<(unsigned)(npairs / 256), 256, 0, stream>>>(Qws, Kws);

    // Causal flash attention (TDM-staged K/V tiles)
    flash_attn<<<dim3(Ss / 64, Hh, Bb), 128, 0, stream>>>(Qws, Kws, Vws, Zws);

    // Output projection (bf16 z -> f32 out, bias fused)
    gemm_wmma<1><<<ggrid, 256, 0, stream>>>(Zws, WO, bO, (float*)d_out);
}
